// RNN_26396869001332
// MI455X (gfx1250) — matrix operand order repaired and checked by reading, not compiled, not run on hardware
//
#include <hip/hip_runtime.h>
#include <hip/hip_bf16.h>

// CDNA5 / gfx1250: wave32, WMMA 16x16x32 bf16 with f32 accumulate.
// f32 GEMMs run as split-bf16 (hi+lo) triple-WMMA for ~f32 accuracy at
// bf16 matrix-core rate (16384 FLOP/inst vs 2048 for V_WMMA_F32_16X16X4_F32).
// K-loop is software-pipelined: next chunk's global float4 loads are issued
// right after the post-store barrier so they overlap the current WMMAs.

typedef __attribute__((ext_vector_type(16))) __bf16 v16bf;
typedef __attribute__((ext_vector_type(8)))  float  v8f;

#define LDP 40   // LDS row pitch in bf16 elems: 32 data + 8 pad (breaks bank conflicts)

#define WMMA_BF16(A, B, C) \
  __builtin_amdgcn_wmma_f32_16x16x32_bf16(false, (A), false, (B), (short)0, (C), false, false)

__device__ __forceinline__ float sigm(float x) { return 1.0f / (1.0f + expf(-x)); }

// Convert 4 f32 -> (hi, lo) bf16 pairs and store to LDS.
__device__ __forceinline__ void stash4(__bf16* __restrict__ hi, __bf16* __restrict__ lo, float4 v) {
  float f0 = v.x, f1 = v.y, f2 = v.z, f3 = v.w;
  __bf16 h0 = (__bf16)f0, h1 = (__bf16)f1, h2 = (__bf16)f2, h3 = (__bf16)f3;
  hi[0] = h0; hi[1] = h1; hi[2] = h2; hi[3] = h3;
  lo[0] = (__bf16)(f0 - (float)h0); lo[1] = (__bf16)(f1 - (float)h1);
  lo[2] = (__bf16)(f2 - (float)h2); lo[3] = (__bf16)(f3 - (float)h3);
}

// A-fragment (16x32 bf16): lane = half*16+l16 holds row M=l16.
// element e<8 -> K = half*8+e ; e>=8 -> K = 16+half*8+(e-8).
__device__ __forceinline__ v16bf load_afrag(const __bf16 (*L)[LDP], int row, int half) {
  v16bf a;
  const __bf16* p0 = &L[row][half * 8];
  const __bf16* p1 = &L[row][16 + half * 8];
#pragma unroll
  for (int e = 0; e < 8; ++e) { a[e] = p0[e]; a[8 + e] = p1[e]; }
  return a;
}

// B-fragment (32x16 bf16): lane holds col N=l16; element e<8 -> K = half*8+e ; e>=8 -> K = 16+half*8+(e-8).
__device__ __forceinline__ v16bf load_bfrag(const __bf16 (*L)[LDP], int row, int half) {
  v16bf b;
  const __bf16* p = &L[row][half * 8];
#pragma unroll
  for (int e = 0; e < 16; ++e) b[e] = p[e < 8 ? e : e + 8];
  return b;
}

// ---------------------------------------------------------------------------
// Embedding gather: out[row][:] = emb[x[row]][:], row = s*B+b, H=1024, 256 thr.
__global__ __launch_bounds__(256) void embed_kernel(const int* __restrict__ x,
                                                    const float* __restrict__ emb,
                                                    float* __restrict__ out, int H) {
  int row = blockIdx.x;
  int tok = x[row];
  const float4* src = (const float4*)&emb[(size_t)tok * H];
  float4* dst = (float4*)&out[(size_t)row * H];
  dst[threadIdx.x] = src[threadIdx.x];
}

// ---------------------------------------------------------------------------
// C[M,N] = A[M,K] @ W[N,K]^T + bias0[N] + bias1[N]   (any bias may be null)
// Block: 128 threads = 4 waves; 64x64 output tile; wave -> 32x32 (2x2 WMMA).
// K staged via LDS in split-bf16 hi/lo form; 12 WMMAs per wave per K=32 chunk.
// Global loads for chunk k+1 are prefetched into registers during chunk k.
__global__ __launch_bounds__(128)
void gemm_awt_bias(const float* __restrict__ A, const float* __restrict__ W,
                   const float* __restrict__ bias0, const float* __restrict__ bias1,
                   float* __restrict__ C, int M, int N, int K) {
  (void)M;
  __shared__ __bf16 lAhi[64][LDP], lAlo[64][LDP], lWhi[64][LDP], lWlo[64][LDP];
  const int t = threadIdx.x;
  const int lane = t & 31, wave = t >> 5;
  const int half = lane >> 4, l16 = lane & 15;
  const int bm = blockIdx.y * 64, bn = blockIdx.x * 64;
  const int wm = (wave >> 1) * 32, wn = (wave & 1) * 32;
  const int srow = t >> 3, scol = (t & 7) * 4;   // staging: 16 rows x 32 cols per pass

  v8f acc[2][2] = {};
  float4 avReg[4], wvReg[4];

  // Prologue: fetch chunk 0.
#pragma unroll
  for (int p = 0; p < 4; ++p) {
    int r = p * 16 + srow;
    avReg[p] = *(const float4*)&A[(size_t)(bm + r) * K + scol];
    wvReg[p] = *(const float4*)&W[(size_t)(bn + r) * K + scol];
  }

  for (int k0 = 0; k0 < K; k0 += 32) {
    // Commit prefetched chunk to LDS (convert f32 -> bf16 hi/lo).
#pragma unroll
    for (int p = 0; p < 4; ++p) {
      int r = p * 16 + srow;
      stash4(&lAhi[r][scol], &lAlo[r][scol], avReg[p]);
      stash4(&lWhi[r][scol], &lWlo[r][scol], wvReg[p]);
    }
    __syncthreads();

    // Issue next chunk's global loads; they overlap the WMMAs below.
    if (k0 + 32 < K) {
#pragma unroll
      for (int p = 0; p < 4; ++p) {
        int r = p * 16 + srow;
        avReg[p] = *(const float4*)&A[(size_t)(bm + r) * K + (k0 + 32) + scol];
        wvReg[p] = *(const float4*)&W[(size_t)(bn + r) * K + (k0 + 32) + scol];
      }
    }

    v16bf ahi[2], alo[2], bhi[2], blo[2];
#pragma unroll
    for (int i = 0; i < 2; ++i) {
      ahi[i] = load_afrag(lAhi, wm + i * 16 + l16, half);
      alo[i] = load_afrag(lAlo, wm + i * 16 + l16, half);
      bhi[i] = load_bfrag(lWhi, wn + i * 16 + l16, half);
      blo[i] = load_bfrag(lWlo, wn + i * 16 + l16, half);
    }
#pragma unroll
    for (int i = 0; i < 2; ++i)
#pragma unroll
      for (int j = 0; j < 2; ++j) {
        acc[i][j] = WMMA_BF16(ahi[i], bhi[j], acc[i][j]);   // hi*hi
        acc[i][j] = WMMA_BF16(ahi[i], blo[j], acc[i][j]);   // hi*lo
        acc[i][j] = WMMA_BF16(alo[i], bhi[j], acc[i][j]);   // lo*hi
      }
    __syncthreads();
  }

  // Epilogue: C/D layout — VGPR r: row = r + 8*half, col = l16.
#pragma unroll
  for (int i = 0; i < 2; ++i)
#pragma unroll
    for (int j = 0; j < 2; ++j) {
      int col = bn + wn + j * 16 + l16;
      float bv = 0.0f;
      if (bias0) bv += bias0[col];
      if (bias1) bv += bias1[col];
      int rbase = bm + wm + i * 16 + half * 8;
#pragma unroll
      for (int r = 0; r < 8; ++r)
        C[(size_t)(rbase + r) * N + col] = acc[i][j][r] + bv;
    }
}

// ---------------------------------------------------------------------------
// One LSTM timestep, fused: gates = xp + hprev @ Whh^T ; c,h update.
// Grid: H/16 blocks (each owns 16 hidden columns across all 4 gates).
// Block: 128 threads = 4 waves; wave g computes gate g's 32(batch)x16 tile.
// Same register-prefetch pipeline on the K loop (latency-critical path).
__global__ __launch_bounds__(128)
void lstm_step(const float* __restrict__ xp,    // [B, 4H] slice at time t
               const float* __restrict__ Whh,   // [4H, H]
               const float* __restrict__ hprev, // [B, H]
               float* __restrict__ c,           // [B, H] in/out (block-private cols)
               float* __restrict__ hout,        // [B, H]  (= ys[t])
               int H) {
  __shared__ __bf16 lHhi[32][LDP], lHlo[32][LDP];
  __shared__ __bf16 lWhi[4][16][LDP], lWlo[4][16][LDP];
  __shared__ float gbuf[4][32][17];
  const int H4 = 4 * H;
  const int t = threadIdx.x;
  const int lane = t & 31, g = t >> 5;          // wave id == gate id (i,f,g,o)
  const int half = lane >> 4, l16 = lane & 15;
  const int hBase = blockIdx.x * 16;
  const int hrow = t >> 3, hcol = (t & 7) * 4;      // h staging coords
  const int wrow = lane >> 3, wcol = (lane & 7) * 4; // W staging coords (per wave)

  v8f acc[2] = {};
  float4 hReg[2], wReg[4];

#pragma unroll
  for (int p = 0; p < 2; ++p)
    hReg[p] = *(const float4*)&hprev[(size_t)(p * 16 + hrow) * H + hcol];
#pragma unroll
  for (int p = 0; p < 4; ++p)
    wReg[p] = *(const float4*)&Whh[(size_t)(g * H + hBase + p * 4 + wrow) * H + wcol];

  for (int k0 = 0; k0 < H; k0 += 32) {
#pragma unroll
    for (int p = 0; p < 2; ++p) {
      int r = p * 16 + hrow;
      stash4(&lHhi[r][hcol], &lHlo[r][hcol], hReg[p]);
    }
#pragma unroll
    for (int p = 0; p < 4; ++p) {
      int r = p * 4 + wrow;
      stash4(&lWhi[g][r][wcol], &lWlo[g][r][wcol], wReg[p]);
    }
    __syncthreads();

    if (k0 + 32 < H) {
#pragma unroll
      for (int p = 0; p < 2; ++p)
        hReg[p] = *(const float4*)&hprev[(size_t)(p * 16 + hrow) * H + (k0 + 32) + hcol];
#pragma unroll
      for (int p = 0; p < 4; ++p)
        wReg[p] = *(const float4*)&Whh[(size_t)(g * H + hBase + p * 4 + wrow) * H + (k0 + 32) + wcol];
    }

    v16bf bhi = load_bfrag(lWhi[g], l16, half);
    v16bf blo = load_bfrag(lWlo[g], l16, half);
#pragma unroll
    for (int i = 0; i < 2; ++i) {               // 2 M-tiles over batch=32
      v16bf ahi = load_afrag(lHhi, i * 16 + l16, half);
      v16bf alo = load_afrag(lHlo, i * 16 + l16, half);
      acc[i] = WMMA_BF16(ahi, bhi, acc[i]);
      acc[i] = WMMA_BF16(ahi, blo, acc[i]);
      acc[i] = WMMA_BF16(alo, bhi, acc[i]);
    }
    __syncthreads();
  }

  // Add input projection, park gates in LDS for cross-wave combine.
#pragma unroll
  for (int i = 0; i < 2; ++i)
#pragma unroll
    for (int r = 0; r < 8; ++r) {
      int b = i * 16 + r + half * 8;
      gbuf[g][b][l16] = acc[i][r] + xp[(size_t)b * H4 + g * H + hBase + l16];
    }
  __syncthreads();

  // Elementwise cell update: 32 batches x 16 cols = 512 elems / 128 threads.
#pragma unroll
  for (int p = 0; p < 4; ++p) {
    int idx = p * 128 + t;
    int b = idx >> 4, hh = idx & 15;
    float iv = sigm(gbuf[0][b][hh]);
    float fv = sigm(gbuf[1][b][hh]);
    float gv = tanhf(gbuf[2][b][hh]);
    float ov = sigm(gbuf[3][b][hh]);
    size_t ci = (size_t)b * H + hBase + hh;
    float cv = fv * c[ci] + iv * gv;
    c[ci] = cv;
    hout[ci] = ov * tanhf(cv);
  }
}

// ---------------------------------------------------------------------------
extern "C" void kernel_launch(void* const* d_in, const int* in_sizes, int n_in,
                              void* d_out, int out_size, void* d_ws, size_t ws_size,
                              hipStream_t stream) {
  (void)in_sizes; (void)n_in; (void)out_size; (void)ws_size;
  constexpr int S = 128, B = 32, H = 1024, L = 2, V = 32000;
  constexpr int H4 = 4 * H;
  constexpr int Mrows = S * B;                    // 4096 sequence*batch rows

  const int*   x    = (const int*)d_in[0];
  const float* h0   = (const float*)d_in[1];      // [L,B,H]
  const float* c0   = (const float*)d_in[2];      // [L,B,H]
  const float* emb  = (const float*)d_in[3];      // [V,H]
  const float* W_ih = (const float*)d_in[4];      // [L,4H,H]
  const float* W_hh = (const float*)d_in[5];      // [L,4H,H]
  const float* b_ih = (const float*)d_in[6];      // [L,4H]
  const float* b_hh = (const float*)d_in[7];      // [L,4H]
  const float* fc_w = (const float*)d_in[8];      // [V,H]
  const float* fc_b = (const float*)d_in[9];      // [V]
  float* out = (float*)d_out;

  // Workspace layout (floats): bufA | bufB | xproj | hb0 | cb   (~96 MB)
  float* ws    = (float*)d_ws;
  float* bufA  = ws;                                   // [S*B, H]
  float* bufB  = bufA + (size_t)Mrows * H;             // [S*B, H]
  float* xproj = bufB + (size_t)Mrows * H;             // [S*B, 4H]
  float* hb0   = xproj + (size_t)Mrows * H4;           // [B, H]
  float* cb    = hb0 + (size_t)B * H;                  // [B, H]

  const size_t logitsN = (size_t)Mrows * V;
  float* outHs = out + logitsN;                        // [L,B,H]
  float* outCs = outHs + (size_t)L * B * H;            // [L,B,H]
  const size_t bhBytes = (size_t)B * H * sizeof(float);

  // 1) Embedding gather -> bufA
  embed_kernel<<<Mrows, 256, 0, stream>>>(x, emb, bufA, H);

  const float* cur = bufA;   // layer input sequence
  float*       nxt = bufB;   // layer output sequence (ys == h_t history)
  for (int l = 0; l < L; ++l) {
    // 2) Input projection for all timesteps: xproj = cur @ W_ih^T + b_ih + b_hh
    gemm_awt_bias<<<dim3(H4 / 64, Mrows / 64), 128, 0, stream>>>(
        cur, W_ih + (size_t)l * H4 * H, b_ih + (size_t)l * H4, b_hh + (size_t)l * H4,
        xproj, Mrows, H4, H);

    // 3) Init recurrent state
    hipMemcpyAsync(hb0, h0 + (size_t)l * B * H, bhBytes, hipMemcpyDeviceToDevice, stream);
    hipMemcpyAsync(cb,  c0 + (size_t)l * B * H, bhBytes, hipMemcpyDeviceToDevice, stream);

    // 4) Sequential recurrence; h_t written straight into ys[t] (= nxt slice)
    for (int t = 0; t < S; ++t) {
      const float* hp = t ? (nxt + (size_t)(t - 1) * B * H) : hb0;
      lstm_step<<<H / 16, 128, 0, stream>>>(
          xproj + (size_t)t * B * H4, W_hh + (size_t)l * H4 * H,
          hp, cb, nxt + (size_t)t * B * H, H);
    }

    // 5) Stash hT, cT for this layer
    hipMemcpyAsync(outHs + (size_t)l * B * H, nxt + (size_t)(S - 1) * B * H, bhBytes,
                   hipMemcpyDeviceToDevice, stream);
    hipMemcpyAsync(outCs + (size_t)l * B * H, cb, bhBytes,
                   hipMemcpyDeviceToDevice, stream);

    float* tmp = (float*)cur; cur = nxt; nxt = tmp;   // ping-pong
  }

  // 6) Final FC: logits = cur @ fc_w^T + fc_b   (N = 32000 = 500 * 64)
  gemm_awt_bias<<<dim3(V / 64, Mrows / 64), 128, 0, stream>>>(
      cur, fc_w, fc_b, nullptr, out, Mrows, V, H);
}
